// MoE_74629351735812
// MI455X (gfx1250) — matrix operand order repaired and checked by reading, not compiled, not run on hardware
//
#include <hip/hip_runtime.h>
#include <hip/hip_bf16.h>
#include <math.h>

typedef __attribute__((ext_vector_type(16))) __bf16 v16bf;
typedef __attribute__((ext_vector_type(8)))  float  v8f;
typedef int v4i __attribute__((ext_vector_type(4)));

constexpr int NTOK = 8192;             // B*S
constexpr int Hdim = 1024;
constexpr int Fdim = 4096;
constexpr int NE   = 8;
constexpr int MT   = 128;              // GEMM tile (M and N)
constexpr int SLOTS_PAD = 17408;       // 16384 + 8*127, rounded to 128
constexpr int MTILES = SLOTS_PAD / MT; // 136

// ---- CDNA5 async global->LDS copy (ASYNCcnt) with compile-safe fallback ----
#if defined(__has_builtin)
#  if __has_builtin(__builtin_amdgcn_global_load_async_to_lds_b128) && \
      __has_builtin(__builtin_amdgcn_s_wait_asynccnt)
#    define MOE_ASYNC 1
#  endif
#endif
#ifndef MOE_ASYNC
#  define MOE_ASYNC 0
#endif

__device__ __forceinline__ void cp16(void* lds, const void* g) {
#if MOE_ASYNC
    __builtin_amdgcn_global_load_async_to_lds_b128(
        (__attribute__((address_space(1))) v4i*)(void*)g,
        (__attribute__((address_space(3))) v4i*)lds, 0, 0);
#else
    *(uint4*)lds = *(const uint4*)g;
#endif
}
__device__ __forceinline__ void cp_wait() {
#if MOE_ASYNC
    __builtin_amdgcn_s_wait_asynccnt(0);
#endif
}

__device__ __forceinline__ unsigned short f2bf(float f) {
    unsigned u = __float_as_uint(f);
    u += 0x7FFFu + ((u >> 16) & 1u);   // round-to-nearest-even
    return (unsigned short)(u >> 16);
}

union Frag { v16bf v; uint4 q[2]; unsigned short u[16]; };
union U8   { uint4 q; unsigned short u[8]; };

// ---------------------------------------------------------------- init
__global__ void moe_init(int* count, int* slotTok) {
    int i = blockIdx.x * blockDim.x + threadIdx.x;
    if (i < NE) count[i] = 0;
    if (i < SLOTS_PAD) slotTok[i] = 0;
}

// ---------------------------------------------------------------- router (wave per token)
__global__ __launch_bounds__(256) void moe_router(
    const float* __restrict__ x, const float* __restrict__ Wr, const float* __restrict__ br,
    float* __restrict__ impPartial, int* __restrict__ count,
    int* __restrict__ tokIdx, float* __restrict__ tokW)
{
    __shared__ float probs[8][NE];
    int wave = threadIdx.x >> 5, lane = threadIdx.x & 31;
    int token = blockIdx.x * 8 + wave;
    const float* xp = x + (size_t)token * Hdim;
    float part[NE];
#pragma unroll
    for (int e = 0; e < NE; ++e) part[e] = 0.f;
    for (int h = lane; h < Hdim; h += 32) {
        float xv = xp[h];
        const float4* wp = (const float4*)(Wr + (size_t)h * NE);
        float4 w0 = wp[0], w1 = wp[1];
        part[0] += xv * w0.x; part[1] += xv * w0.y;
        part[2] += xv * w0.z; part[3] += xv * w0.w;
        part[4] += xv * w1.x; part[5] += xv * w1.y;
        part[6] += xv * w1.z; part[7] += xv * w1.w;
    }
#pragma unroll
    for (int e = 0; e < NE; ++e)
#pragma unroll
        for (int off = 16; off > 0; off >>= 1)
            part[e] += __shfl_xor(part[e], off, 32);
    if (lane == 0) {
        float lg[NE], mx = -1e30f;
#pragma unroll
        for (int e = 0; e < NE; ++e) { lg[e] = part[e] + br[e]; mx = fmaxf(mx, lg[e]); }
        float s = 0.f, p[NE];
#pragma unroll
        for (int e = 0; e < NE; ++e) { p[e] = __expf(lg[e] - mx); s += p[e]; }
        float inv = 1.f / s;
#pragma unroll
        for (int e = 0; e < NE; ++e) probs[wave][e] = p[e] * inv;
        int i0 = 0;
#pragma unroll
        for (int e = 1; e < NE; ++e) if (lg[e] > lg[i0]) i0 = e;
        int i1 = (i0 == 0) ? 1 : 0;
#pragma unroll
        for (int e = 0; e < NE; ++e) if (e != i0 && lg[e] > lg[i1]) i1 = e;
        float e1 = __expf(lg[i1] - lg[i0]);
        float w0 = 1.f / (1.f + e1);
        tokIdx[token * 2] = i0; tokIdx[token * 2 + 1] = i1;
        tokW[token * 2] = w0;  tokW[token * 2 + 1] = e1 * w0;
        atomicAdd(&count[i0], 1); atomicAdd(&count[i1], 1);
    }
    __syncthreads();
    if (threadIdx.x < NE) {
        float s = 0.f;
#pragma unroll
        for (int w = 0; w < 8; ++w) s += probs[w][threadIdx.x];
        impPartial[blockIdx.x * NE + threadIdx.x] = s;
    }
}

// ---------------------------------------------------------------- aux loss + segment offsets
__global__ void moe_finalize(const float* __restrict__ impPartial, const int* __restrict__ count,
                             int* __restrict__ offsets, int* __restrict__ cursor,
                             float* __restrict__ auxOut, int nPart)
{
    __shared__ float imp[NE];
    int t = threadIdx.x;
    if (t < NE) {
        float s = 0.f;
        for (int b = 0; b < nPart; ++b) s += impPartial[b * NE + t];
        imp[t] = s;
    }
    __syncthreads();
    if (t == 0) {
        float a = 0.f;
        for (int e = 0; e < NE; ++e) { float v = imp[e] / (float)NTOK; a += v * v; }
        auxOut[0] = (float)NE * a;
        int off = 0;
        for (int e = 0; e < NE; ++e) {
            offsets[e] = off; cursor[e] = off;
            off += ((count[e] + MT - 1) / MT) * MT;
        }
        offsets[NE] = off;
    }
}

// ---------------------------------------------------------------- scatter tokens -> slots
__global__ void moe_scatter(const int* __restrict__ tokIdx, int* __restrict__ cursor,
                            int* __restrict__ slotTok, int* __restrict__ tokSlot)
{
    int t = blockIdx.x * blockDim.x + threadIdx.x;
    if (t >= NTOK) return;
#pragma unroll
    for (int k = 0; k < 2; ++k) {
        int e = tokIdx[t * 2 + k];
        int pos = atomicAdd(&cursor[e], 1);
        slotTok[pos] = t;
        tokSlot[t * 2 + k] = pos;
    }
}

// ---------------------------------------------------------------- fp32 -> bf16 pre-conversion
__global__ void moe_cvt_x(const float* __restrict__ x, unsigned short* __restrict__ xbf) {
    size_t i = ((size_t)blockIdx.x * blockDim.x + threadIdx.x) * 8;
    const float4* p = (const float4*)(x + i);
    float4 a = p[0], b = p[1];
    U8 o;
    o.u[0]=f2bf(a.x); o.u[1]=f2bf(a.y); o.u[2]=f2bf(a.z); o.u[3]=f2bf(a.w);
    o.u[4]=f2bf(b.x); o.u[5]=f2bf(b.y); o.u[6]=f2bf(b.z); o.u[7]=f2bf(b.w);
    *(uint4*)(xbf + i) = o.q;
}

// Convert W (E x Ktot x Ntot, fp32) into per-(expert, n-tile, k-step) 8KB bf16 blobs whose
// byte image equals the LDS B-fragment image consumed by the GEMM (so GEMM copies are raw b128).
__global__ __launch_bounds__(256) void moe_cvt_w(const float* __restrict__ W,
                                                 unsigned short* __restrict__ Wbf,
                                                 int Ntot, int ksPerN)
{
    int e = blockIdx.y;
    int nt = blockIdx.x / ksPerN, ks = blockIdx.x % ksPerN;
    int tid = threadIdx.x, lane = tid & 31, lane15 = lane & 15, half = lane >> 4;
    int k0 = ks * 32, n0 = nt * MT;
    size_t Ktot = (size_t)ksPerN * 32;
    const float* src = W + (size_t)e * Ktot * Ntot
                         + (size_t)(k0 + half * 8) * Ntot
                         + (n0 + (tid >> 5) * 16 + lane15);
    Frag f;
#pragma unroll
    for (int j = 0; j < 16; ++j) f.u[j] = f2bf(src[(size_t)(j + (j & 8)) * Ntot]);
    unsigned short* dst = Wbf + ((size_t)e * gridDim.x + blockIdx.x) * 4096 + (size_t)tid * 16;
    *(uint4*)dst       = f.q[0];
    *(uint4*)(dst + 8) = f.q[1];
}

// ---------------------------------------------------------------- pre-converted GEMM1
// h = GELU(x W1 + b1) -> bf16 ; async double-buffered LDS, 1 barrier / K-step
__global__ __launch_bounds__(256) void moe_ffn1_pre(
    const unsigned short* __restrict__ xbf, const unsigned short* __restrict__ W1bf,
    const float* __restrict__ b1, const int* __restrict__ slotTok,
    const int* __restrict__ offsets, unsigned short* __restrict__ hbuf)
{
    __shared__ unsigned short sA[2][8 * 512];
    __shared__ unsigned short sB[2][8 * 512];
    int m0 = blockIdx.y * MT;
    int off[NE + 1];
#pragma unroll
    for (int i = 0; i <= NE; ++i) off[i] = offsets[i];
    if (m0 >= off[NE]) return;
    int e = 0;
#pragma unroll
    for (int i = 0; i < NE; ++i) if (m0 >= off[i + 1]) e = i + 1;
    int n0 = blockIdx.x * MT;
    int tid = threadIdx.x, wave = tid >> 5, lane = tid & 31;
    int lane15 = lane & 15, half = lane >> 4;
    int token = slotTok[m0 + wave * 16 + lane15];
    const unsigned short* arow  = xbf + (size_t)token * Hdim;
    const unsigned short* bblob = W1bf + ((size_t)e * 1024 + (size_t)(n0 / MT) * (Hdim / 32)) * 4096;
    const int KS = Hdim / 32;
    int mw = wave >> 1, nw = wave & 1;
    v8f acc[2][4];
#pragma unroll
    for (int i = 0; i < 2; ++i)
#pragma unroll
        for (int j = 0; j < 4; ++j) acc[i][j] = (v8f){0,0,0,0,0,0,0,0};

    {   // prologue: tile ks=0 -> buffer 0
        const unsigned short* ap = arow + half * 8;
        cp16(&sA[0][tid * 16],     ap);
        cp16(&sA[0][tid * 16 + 8], ap + 16);
        const unsigned short* bp = bblob + (size_t)tid * 16;
        cp16(&sB[0][tid * 16],     bp);
        cp16(&sB[0][tid * 16 + 8], bp + 8);
    }
    for (int ks = 0; ks < KS; ++ks) {
        int cur = ks & 1;
        cp_wait();
        __syncthreads();
        if (ks + 1 < KS) {
            int nxt = cur ^ 1;
            const unsigned short* ap = arow + (ks + 1) * 32 + half * 8;
            cp16(&sA[nxt][tid * 16],     ap);
            cp16(&sA[nxt][tid * 16 + 8], ap + 16);
            const unsigned short* bp = bblob + (size_t)(ks + 1) * 4096 + (size_t)tid * 16;
            cp16(&sB[nxt][tid * 16],     bp);
            cp16(&sB[nxt][tid * 16 + 8], bp + 8);
        }
        Frag Af[2], Bf[4];
#pragma unroll
        for (int i = 0; i < 2; ++i) {
            const uint4* p = (const uint4*)(&sA[cur][(2 * mw + i) * 512 + lane * 16]);
            Af[i].q[0] = p[0]; Af[i].q[1] = p[1];
        }
#pragma unroll
        for (int j = 0; j < 4; ++j) {
            const uint4* p = (const uint4*)(&sB[cur][(4 * nw + j) * 512 + lane * 16]);
            Bf[j].q[0] = p[0]; Bf[j].q[1] = p[1];
        }
#pragma unroll
        for (int i = 0; i < 2; ++i)
#pragma unroll
            for (int j = 0; j < 4; ++j)
                acc[i][j] = __builtin_amdgcn_wmma_f32_16x16x32_bf16(
                    false, Af[i].v, false, Bf[j].v, (short)0, acc[i][j], false, false);
    }
    const float* b1e = b1 + (size_t)e * Fdim;
#pragma unroll
    for (int i = 0; i < 2; ++i)
#pragma unroll
        for (int j = 0; j < 4; ++j) {
            int mbase = m0 + (2 * mw + i) * 16 + half * 8;
            int f = n0 + (4 * nw + j) * 16 + lane15;
            float bias = b1e[f];
#pragma unroll
            for (int r = 0; r < 8; ++r) {
                float v = acc[i][j][r] + bias;
                v = 0.5f * v * (1.f + erff(v * 0.70710678118654752f));  // exact GELU
                hbuf[(size_t)(mbase + r) * Fdim + f] = f2bf(v);
            }
        }
}

// ---------------------------------------------------------------- pre-converted GEMM2
__global__ __launch_bounds__(256) void moe_ffn2_pre(
    const unsigned short* __restrict__ hbuf, const unsigned short* __restrict__ W2bf,
    const float* __restrict__ b2, const int* __restrict__ offsets, float* __restrict__ ybuf)
{
    __shared__ unsigned short sA[2][8 * 512];
    __shared__ unsigned short sB[2][8 * 512];
    int m0 = blockIdx.y * MT;
    int off[NE + 1];
#pragma unroll
    for (int i = 0; i <= NE; ++i) off[i] = offsets[i];
    if (m0 >= off[NE]) return;
    int e = 0;
#pragma unroll
    for (int i = 0; i < NE; ++i) if (m0 >= off[i + 1]) e = i + 1;
    int n0 = blockIdx.x * MT;
    int tid = threadIdx.x, wave = tid >> 5, lane = tid & 31;
    int lane15 = lane & 15, half = lane >> 4;
    const unsigned short* arow  = hbuf + (size_t)(m0 + wave * 16 + lane15) * Fdim;
    const unsigned short* bblob = W2bf + ((size_t)e * 1024 + (size_t)(n0 / MT) * (Fdim / 32)) * 4096;
    const int KS = Fdim / 32;
    int mw = wave >> 1, nw = wave & 1;
    v8f acc[2][4];
#pragma unroll
    for (int i = 0; i < 2; ++i)
#pragma unroll
        for (int j = 0; j < 4; ++j) acc[i][j] = (v8f){0,0,0,0,0,0,0,0};

    {
        const unsigned short* ap = arow + half * 8;
        cp16(&sA[0][tid * 16],     ap);
        cp16(&sA[0][tid * 16 + 8], ap + 16);
        const unsigned short* bp = bblob + (size_t)tid * 16;
        cp16(&sB[0][tid * 16],     bp);
        cp16(&sB[0][tid * 16 + 8], bp + 8);
    }
    for (int ks = 0; ks < KS; ++ks) {
        int cur = ks & 1;
        cp_wait();
        __syncthreads();
        if (ks + 1 < KS) {
            int nxt = cur ^ 1;
            const unsigned short* ap = arow + (ks + 1) * 32 + half * 8;
            cp16(&sA[nxt][tid * 16],     ap);
            cp16(&sA[nxt][tid * 16 + 8], ap + 16);
            const unsigned short* bp = bblob + (size_t)(ks + 1) * 4096 + (size_t)tid * 16;
            cp16(&sB[nxt][tid * 16],     bp);
            cp16(&sB[nxt][tid * 16 + 8], bp + 8);
        }
        Frag Af[2], Bf[4];
#pragma unroll
        for (int i = 0; i < 2; ++i) {
            const uint4* p = (const uint4*)(&sA[cur][(2 * mw + i) * 512 + lane * 16]);
            Af[i].q[0] = p[0]; Af[i].q[1] = p[1];
        }
#pragma unroll
        for (int j = 0; j < 4; ++j) {
            const uint4* p = (const uint4*)(&sB[cur][(4 * nw + j) * 512 + lane * 16]);
            Bf[j].q[0] = p[0]; Bf[j].q[1] = p[1];
        }
#pragma unroll
        for (int i = 0; i < 2; ++i)
#pragma unroll
            for (int j = 0; j < 4; ++j)
                acc[i][j] = __builtin_amdgcn_wmma_f32_16x16x32_bf16(
                    false, Af[i].v, false, Bf[j].v, (short)0, acc[i][j], false, false);
    }
    const float* b2e = b2 + (size_t)e * Hdim;
#pragma unroll
    for (int i = 0; i < 2; ++i)
#pragma unroll
        for (int j = 0; j < 4; ++j) {
            int mbase = m0 + (2 * mw + i) * 16 + half * 8;
            int h = n0 + (4 * nw + j) * 16 + lane15;
            float bias = b2e[h];
#pragma unroll
            for (int r = 0; r < 8; ++r)
                ybuf[(size_t)(mbase + r) * Hdim + h] = acc[i][j][r] + bias;
        }
}

// ---------------------------------------------------------------- fallback GEMMs (fp32 in, convert in-loop)
__global__ __launch_bounds__(256) void moe_ffn1_fb(
    const float* __restrict__ x, const float* __restrict__ W1, const float* __restrict__ b1,
    const int* __restrict__ slotTok, const int* __restrict__ offsets,
    unsigned short* __restrict__ hbuf)
{
    __shared__ unsigned short sA[8 * 512];
    __shared__ unsigned short sB[8 * 512];
    int m0 = blockIdx.y * MT;
    int off[NE + 1];
#pragma unroll
    for (int i = 0; i <= NE; ++i) off[i] = offsets[i];
    if (m0 >= off[NE]) return;
    int e = 0;
#pragma unroll
    for (int i = 0; i < NE; ++i) if (m0 >= off[i + 1]) e = i + 1;
    int n0 = blockIdx.x * MT;
    int tid = threadIdx.x, wave = tid >> 5, lane = tid & 31;
    int lane15 = lane & 15, half = lane >> 4;
    int token = slotTok[m0 + wave * 16 + lane15];
    const float* xrow = x + (size_t)token * Hdim;
    const float* W1e  = W1 + (size_t)e * Hdim * Fdim;
    int mw = wave >> 1, nw = wave & 1;
    v8f acc[2][4];
#pragma unroll
    for (int i = 0; i < 2; ++i)
#pragma unroll
        for (int j = 0; j < 4; ++j) acc[i][j] = (v8f){0,0,0,0,0,0,0,0};
    for (int k0 = 0; k0 < Hdim; k0 += 32) {
        const float* ap = xrow + k0 + half * 8;
        float4 a0 = *(const float4*)ap,        a1 = *(const float4*)(ap + 4);
        float4 a2 = *(const float4*)(ap + 16), a3 = *(const float4*)(ap + 20);
        Frag fa;
        fa.u[0]=f2bf(a0.x); fa.u[1]=f2bf(a0.y); fa.u[2]=f2bf(a0.z); fa.u[3]=f2bf(a0.w);
        fa.u[4]=f2bf(a1.x); fa.u[5]=f2bf(a1.y); fa.u[6]=f2bf(a1.z); fa.u[7]=f2bf(a1.w);
        fa.u[8]=f2bf(a2.x); fa.u[9]=f2bf(a2.y); fa.u[10]=f2bf(a2.z); fa.u[11]=f2bf(a2.w);
        fa.u[12]=f2bf(a3.x); fa.u[13]=f2bf(a3.y); fa.u[14]=f2bf(a3.z); fa.u[15]=f2bf(a3.w);
        *(uint4*)(sA + wave * 512 + lane * 16)     = fa.q[0];
        *(uint4*)(sA + wave * 512 + lane * 16 + 8) = fa.q[1];
        const float* bp = W1e + (size_t)(k0 + half * 8) * Fdim + (n0 + wave * 16 + lane15);
        Frag fb;
#pragma unroll
        for (int j = 0; j < 16; ++j) fb.u[j] = f2bf(bp[(size_t)(j + (j & 8)) * Fdim]);
        *(uint4*)(sB + wave * 512 + lane * 16)     = fb.q[0];
        *(uint4*)(sB + wave * 512 + lane * 16 + 8) = fb.q[1];
        __syncthreads();
        Frag Af[2], Bf[4];
#pragma unroll
        for (int i = 0; i < 2; ++i) {
            const uint4* p = (const uint4*)(sA + (2 * mw + i) * 512 + lane * 16);
            Af[i].q[0] = p[0]; Af[i].q[1] = p[1];
        }
#pragma unroll
        for (int j = 0; j < 4; ++j) {
            const uint4* p = (const uint4*)(sB + (4 * nw + j) * 512 + lane * 16);
            Bf[j].q[0] = p[0]; Bf[j].q[1] = p[1];
        }
#pragma unroll
        for (int i = 0; i < 2; ++i)
#pragma unroll
            for (int j = 0; j < 4; ++j)
                acc[i][j] = __builtin_amdgcn_wmma_f32_16x16x32_bf16(
                    false, Af[i].v, false, Bf[j].v, (short)0, acc[i][j], false, false);
        __syncthreads();
    }
    const float* b1e = b1 + (size_t)e * Fdim;
#pragma unroll
    for (int i = 0; i < 2; ++i)
#pragma unroll
        for (int j = 0; j < 4; ++j) {
            int mbase = m0 + (2 * mw + i) * 16 + half * 8;
            int f = n0 + (4 * nw + j) * 16 + lane15;
            float bias = b1e[f];
#pragma unroll
            for (int r = 0; r < 8; ++r) {
                float v = acc[i][j][r] + bias;
                v = 0.5f * v * (1.f + erff(v * 0.70710678118654752f));
                hbuf[(size_t)(mbase + r) * Fdim + f] = f2bf(v);
            }
        }
}

__global__ __launch_bounds__(256) void moe_ffn2_fb(
    const unsigned short* __restrict__ hbuf, const float* __restrict__ W2,
    const float* __restrict__ b2, const int* __restrict__ offsets, float* __restrict__ ybuf)
{
    __shared__ unsigned short sA[8 * 512];
    __shared__ unsigned short sB[8 * 512];
    int m0 = blockIdx.y * MT;
    int off[NE + 1];
#pragma unroll
    for (int i = 0; i <= NE; ++i) off[i] = offsets[i];
    if (m0 >= off[NE]) return;
    int e = 0;
#pragma unroll
    for (int i = 0; i < NE; ++i) if (m0 >= off[i + 1]) e = i + 1;
    int n0 = blockIdx.x * MT;
    int tid = threadIdx.x, wave = tid >> 5, lane = tid & 31;
    int lane15 = lane & 15, half = lane >> 4;
    const unsigned short* hrow = hbuf + (size_t)(m0 + wave * 16 + lane15) * Fdim;
    const float* W2e = W2 + (size_t)e * Fdim * Hdim;
    int mw = wave >> 1, nw = wave & 1;
    v8f acc[2][4];
#pragma unroll
    for (int i = 0; i < 2; ++i)
#pragma unroll
        for (int j = 0; j < 4; ++j) acc[i][j] = (v8f){0,0,0,0,0,0,0,0};
    for (int k0 = 0; k0 < Fdim; k0 += 32) {
        const unsigned short* ap = hrow + k0 + half * 8;
        uint4 q0 = *(const uint4*)ap;
        uint4 q1 = *(const uint4*)(ap + 16);
        *(uint4*)(sA + wave * 512 + lane * 16)     = q0;
        *(uint4*)(sA + wave * 512 + lane * 16 + 8) = q1;
        const float* bp = W2e + (size_t)(k0 + half * 8) * Hdim + (n0 + wave * 16 + lane15);
        Frag fb;
#pragma unroll
        for (int j = 0; j < 16; ++j) fb.u[j] = f2bf(bp[(size_t)(j + (j & 8)) * Hdim]);
        *(uint4*)(sB + wave * 512 + lane * 16)     = fb.q[0];
        *(uint4*)(sB + wave * 512 + lane * 16 + 8) = fb.q[1];
        __syncthreads();
        Frag Af[2], Bf[4];
#pragma unroll
        for (int i = 0; i < 2; ++i) {
            const uint4* p = (const uint4*)(sA + (2 * mw + i) * 512 + lane * 16);
            Af[i].q[0] = p[0]; Af[i].q[1] = p[1];
        }
#pragma unroll
        for (int j = 0; j < 4; ++j) {
            const uint4* p = (const uint4*)(sB + (4 * nw + j) * 512 + lane * 16);
            Bf[j].q[0] = p[0]; Bf[j].q[1] = p[1];
        }
#pragma unroll
        for (int i = 0; i < 2; ++i)
#pragma unroll
            for (int j = 0; j < 4; ++j)
                acc[i][j] = __builtin_amdgcn_wmma_f32_16x16x32_bf16(
                    false, Af[i].v, false, Bf[j].v, (short)0, acc[i][j], false, false);
        __syncthreads();
    }
    const float* b2e = b2 + (size_t)e * Hdim;
#pragma unroll
    for (int i = 0; i < 2; ++i)
#pragma unroll
        for (int j = 0; j < 4; ++j) {
            int mbase = m0 + (2 * mw + i) * 16 + half * 8;
            int h = n0 + (4 * nw + j) * 16 + lane15;
            float bias = b2e[h];
#pragma unroll
            for (int r = 0; r < 8; ++r)
                ybuf[(size_t)(mbase + r) * Hdim + h] = acc[i][j][r] + bias;
        }
}

// ---------------------------------------------------------------- weighted combine
__global__ void moe_combine(const float* __restrict__ ybuf, const int* __restrict__ tokSlot,
                            const float* __restrict__ tokW, float* __restrict__ out)
{
    int t = blockIdx.y;
    int h = blockIdx.x * blockDim.x + threadIdx.x;
    int s0 = tokSlot[t * 2], s1 = tokSlot[t * 2 + 1];
    float w0 = tokW[t * 2], w1 = tokW[t * 2 + 1];
    out[(size_t)t * Hdim + h] =
        w0 * ybuf[(size_t)s0 * Hdim + h] + w1 * ybuf[(size_t)s1 * Hdim + h];
}

// ---------------------------------------------------------------- launcher
extern "C" void kernel_launch(void* const* d_in, const int* in_sizes, int n_in,
                              void* d_out, int out_size, void* d_ws, size_t ws_size,
                              hipStream_t stream)
{
    const float* x  = (const float*)d_in[0];
    const float* W1 = (const float*)d_in[1];
    const float* b1 = (const float*)d_in[2];
    const float* W2 = (const float*)d_in[3];
    const float* b2 = (const float*)d_in[4];
    const float* Wr = (const float*)d_in[5];
    const float* br = (const float*)d_in[6];
    float* out = (float*)d_out;

    const int RB = NTOK / 8;
    char* ws = (char*)d_ws;
    size_t o = 0;
    float* impPartial = (float*)(ws + o); o += (size_t)RB * NE * 4;
    int*   count      = (int*)(ws + o);   o += 64;
    int*   offsets    = (int*)(ws + o);   o += 64;
    int*   cursor     = (int*)(ws + o);   o += 64;
    int*   tokIdx     = (int*)(ws + o);   o += (size_t)NTOK * 2 * 4;
    float* tokW       = (float*)(ws + o); o += (size_t)NTOK * 2 * 4;
    int*   tokSlot    = (int*)(ws + o);   o += (size_t)NTOK * 2 * 4;
    int*   slotTok    = (int*)(ws + o);   o += (size_t)SLOTS_PAD * 4;
    o = (o + 255) & ~(size_t)255;
    unsigned short* hbuf = (unsigned short*)(ws + o); o += (size_t)SLOTS_PAD * Fdim * 2;
    float* ybuf = (float*)(ws + o);                   o += (size_t)SLOTS_PAD * Hdim * 4;
    // extra buffers for the pre-converted fast path
    unsigned short* xbf  = (unsigned short*)(ws + o); o += (size_t)NTOK * Hdim * 2;
    unsigned short* W1bf = (unsigned short*)(ws + o); o += (size_t)NE * Hdim * Fdim * 2;
    unsigned short* W2bf = (unsigned short*)(ws + o); o += (size_t)NE * Fdim * Hdim * 2;
    bool pre = (ws_size >= o);

    moe_init<<<(SLOTS_PAD + 255) / 256, 256, 0, stream>>>(count, slotTok);
    moe_router<<<RB, 256, 0, stream>>>(x, Wr, br, impPartial, count, tokIdx, tokW);
    moe_finalize<<<1, 256, 0, stream>>>(impPartial, count, offsets, cursor,
                                        out + (size_t)NTOK * Hdim, RB);
    moe_scatter<<<NTOK / 256, 256, 0, stream>>>(tokIdx, cursor, slotTok, tokSlot);
    if (pre) {
        moe_cvt_x<<<(NTOK * Hdim) / (256 * 8), 256, 0, stream>>>(x, xbf);
        moe_cvt_w<<<dim3((Fdim / MT) * (Hdim / 32), NE), 256, 0, stream>>>(W1, W1bf, Fdim, Hdim / 32);
        moe_cvt_w<<<dim3((Hdim / MT) * (Fdim / 32), NE), 256, 0, stream>>>(W2, W2bf, Hdim, Fdim / 32);
        moe_ffn1_pre<<<dim3(Fdim / MT, MTILES), 256, 0, stream>>>(xbf, W1bf, b1, slotTok, offsets, hbuf);
        moe_ffn2_pre<<<dim3(Hdim / MT, MTILES), 256, 0, stream>>>(hbuf, W2bf, b2, offsets, ybuf);
    } else {
        moe_ffn1_fb<<<dim3(Fdim / MT, MTILES), 256, 0, stream>>>(x, W1, b1, slotTok, offsets, hbuf);
        moe_ffn2_fb<<<dim3(Hdim / MT, MTILES), 256, 0, stream>>>(hbuf, W2, b2, offsets, ybuf);
    }
    moe_combine<<<dim3(Hdim / 256, NTOK), 256, 0, stream>>>(ybuf, tokSlot, tokW, out);
}
